// GCN_18236431139460
// MI455X (gfx1250) — matrix operand order repaired and checked by reading, not compiled, not run on hardware
//
#include <hip/hip_runtime.h>
#include <hip/hip_bf16.h>

typedef __attribute__((ext_vector_type(16))) _Float16 v16h;
typedef __attribute__((ext_vector_type(8)))  _Float16 v8h;
typedef __attribute__((ext_vector_type(4)))  _Float16 v4h;
typedef __attribute__((ext_vector_type(8)))  float    v8f;
typedef __attribute__((ext_vector_type(4)))  float    v4f;

#define N_TOT        32768
#define NUM_NODE     128
#define B_GRAPHS     256
#define FDIM         128
#define EDGES_PER_G  2048          // NUM_NODE * DEG
#define MLP_H        512
#define OUT_DIM      10
#define KDIM_MLP     16384         // NUM_NODE * FDIM

#define KCHUNK       128           // K staged per LDS refill
#define NBLK         64            // N columns per block
#define LPAD         136           // padded LDS k-stride in halves: 272B, 16B-aligned

// ---------------- degree / normalization ----------------
__global__ void k_deg_init(float* deg) {
    int i = blockIdx.x * blockDim.x + threadIdx.x;
    if (i < N_TOT) deg[i] = 1.0f;                    // self-loop contribution
}
__global__ void k_deg_accum(const int* dst, float* deg, int E) {
    int i = blockIdx.x * blockDim.x + threadIdx.x;
    if (i < E) atomicAdd(&deg[dst[i]], 1.0f);
}
__global__ void k_rsqrt(float* d) {
    int i = blockIdx.x * blockDim.x + threadIdx.x;
    if (i < N_TOT) d[i] = rsqrtf(d[i]);              // deg >= 1 always
}

__device__ __forceinline__ v4h cvt4(v4f v) {
    v4h r;
#pragma unroll
    for (int j = 0; j < 4; ++j) r[j] = (_Float16)v[j];
    return r;
}
#define CAT16(lo, hi) __builtin_shufflevector(lo, hi, 0,1,2,3,4,5,6,7,8,9,10,11,12,13,14,15)

// ---------------- WMMA GEMM: C[M,N] = A[M,K] @ B[K,N] (+bias)(+relu) ----------------
// Block = WAVES waves, each computing a 16x64 tile (4 accumulators).
// Both operands staged in LDS as f16 (converted once at staging), so the inner
// loop is pure ds_load_b128 -> v_wmma_f32_16x16x32_f16 with no conversions.
// Requires: K % 128 == 0, N % 64 == 0, M % (16*WAVES) == 0.
template<int WAVES, bool BIAS, bool RELU>
__global__ void k_wmma_gemm(const float* __restrict__ A, const float* __restrict__ Bm,
                            const float* __restrict__ bias, float* __restrict__ C,
                            int M, int N, int K) {
    __shared__ _Float16 ash[16 * WAVES * LPAD];      // A chunk  [row][k]
    __shared__ _Float16 bsh[NBLK * LPAD];            // B chunk  [n][k] (transposed)

    const int nbn  = N / NBLK;
    const int bn   = blockIdx.x % nbn;
    const int bm   = blockIdx.x / nbn;
    const int n0   = bn * NBLK;
    const int w    = threadIdx.x >> 5;
    const int lane = threadIdx.x & 31;
    const int h    = lane >> 4;
    const int lo   = lane & 15;
    const int ROWS = 16 * WAVES;
    const int mrow = bm * ROWS;                      // block's output row base

    v8f acc0 = {}, acc1 = {}, acc2 = {}, acc3 = {};

    const int aoff  = (w * 16 + lo) * LPAD;          // this lane's A row in LDS
    const int boff0 = lo * LPAD;                     // this lane's B rows in LDS

    for (int kc = 0; kc < K; kc += KCHUNK) {
        if (kc) __syncthreads();
        // ---- stage A[mrow:mrow+ROWS, kc:kc+128] as f16 (b128 load, b64 store) ----
        for (int i = threadIdx.x; i < ROWS * (KCHUNK / 4); i += WAVES * 32) {
            int row = i >> 5;                        // 0..ROWS-1
            int kq  = i & 31;                        // float4 index along k
            v4f v   = *(const v4f*)(A + (size_t)(mrow + row) * K + kc + kq * 4);
            *(v4h*)(&ash[row * LPAD + kq * 4]) = cvt4(v);
        }
        // ---- stage B[kc:kc+128, n0:n0+64] as f16, transposed ----
        for (int i = threadIdx.x; i < (KCHUNK * NBLK) / 4; i += WAVES * 32) {
            int kl = i >> 4;                         // 0..127
            int nq = i & 15;                         // float4 within row
            v4f v  = *(const v4f*)(Bm + (size_t)(kc + kl) * N + n0 + nq * 4);
            v4h hv = cvt4(v);
#pragma unroll
            for (int j = 0; j < 4; ++j)
                bsh[(nq * 4 + j) * LPAD + kl] = hv[j];
        }
        __syncthreads();

        // ---- 4 k-steps of 32: load all fragments first, then 4 WMMAs ----
#pragma unroll
        for (int k0 = 0; k0 < KCHUNK; k0 += 32) {
            v8h alo = *(const v8h*)(&ash[aoff + k0 + 16 * h]);
            v8h ahi = *(const v8h*)(&ash[aoff + k0 + 16 * h + 8]);
            const int kk = k0 + 16 * h;
            v8h b0l = *(const v8h*)(&bsh[boff0 + 0 * 16 * LPAD + kk]);
            v8h b0h = *(const v8h*)(&bsh[boff0 + 0 * 16 * LPAD + kk + 8]);
            v8h b1l = *(const v8h*)(&bsh[boff0 + 1 * 16 * LPAD + kk]);
            v8h b1h = *(const v8h*)(&bsh[boff0 + 1 * 16 * LPAD + kk + 8]);
            v8h b2l = *(const v8h*)(&bsh[boff0 + 2 * 16 * LPAD + kk]);
            v8h b2h = *(const v8h*)(&bsh[boff0 + 2 * 16 * LPAD + kk + 8]);
            v8h b3l = *(const v8h*)(&bsh[boff0 + 3 * 16 * LPAD + kk]);
            v8h b3h = *(const v8h*)(&bsh[boff0 + 3 * 16 * LPAD + kk + 8]);
            v16h a  = CAT16(alo, ahi);
            acc0 = __builtin_amdgcn_wmma_f32_16x16x32_f16(false, a, false, CAT16(b0l, b0h),
                                                          (short)0, acc0, false, false);
            acc1 = __builtin_amdgcn_wmma_f32_16x16x32_f16(false, a, false, CAT16(b1l, b1h),
                                                          (short)0, acc1, false, false);
            acc2 = __builtin_amdgcn_wmma_f32_16x16x32_f16(false, a, false, CAT16(b2l, b2h),
                                                          (short)0, acc2, false, false);
            acc3 = __builtin_amdgcn_wmma_f32_16x16x32_f16(false, a, false, CAT16(b3l, b3h),
                                                          (short)0, acc3, false, false);
        }
    }

    // ---- writeout: D VGPR v -> row = v + 8h, col = lo (+16 per n-subtile) ----
    const int myrow = mrow + w * 16;
    v8f* accs[4] = { &acc0, &acc1, &acc2, &acc3 };
#pragma unroll
    for (int t = 0; t < 4; ++t) {
        int col = n0 + t * 16 + lo;
        float bv = BIAS ? bias[col] : 0.0f;
#pragma unroll
        for (int v = 0; v < 8; ++v) {
            int   row = myrow + v + 8 * h;
            float val = (*accs[t])[v] + bv;
            if (RELU) val = fmaxf(val, 0.0f);
            C[(size_t)row * N + col] = val;
        }
    }
}

// ---------------- per-graph GCN aggregation, fully LDS-resident ----------------
__global__ void k_gcn_agg(const int* __restrict__ src, const int* __restrict__ dst,
                          const float* __restrict__ dinv, const float* __restrict__ hin,
                          const float* __restrict__ bias, float* __restrict__ out) {
    __shared__ float lagg[NUM_NODE * FDIM];          // 64 KB
    __shared__ float lh[NUM_NODE * FDIM];            // 64 KB
    __shared__ int   lsrc[EDGES_PER_G];              // 8 KB
    __shared__ int   ldst[EDGES_PER_G];              // 8 KB
    __shared__ float ldinv[NUM_NODE];                // 512 B

    const int g    = blockIdx.x;
    const int base = g * NUM_NODE;
    const int e0   = g * EDGES_PER_G;
    const int tid  = threadIdx.x;

    for (int i = tid; i < NUM_NODE * FDIM; i += blockDim.x) {
        lagg[i] = 0.0f;
        lh[i]   = hin[(size_t)base * FDIM + i];
    }
    for (int i = tid; i < EDGES_PER_G; i += blockDim.x) {
        lsrc[i] = src[e0 + i] - base;                // localize node ids
        ldst[i] = dst[e0 + i] - base;
    }
    if (tid < NUM_NODE) ldinv[tid] = dinv[base + tid];
    __syncthreads();

    const int f = tid & (FDIM - 1);                  // feature lane (contiguous per wave)
    for (int i = tid >> 7; i < EDGES_PER_G; i += (blockDim.x >> 7)) {
        int   r    = lsrc[i];
        int   c    = ldst[i];
        float norm = ldinv[r] * ldinv[c];
        atomicAdd(&lagg[c * FDIM + f], lh[r * FDIM + f] * norm);
    }
    __syncthreads();

    for (int i = tid; i < NUM_NODE * FDIM; i += blockDim.x) {
        int   node = i >> 7;
        int   ff   = i & (FDIM - 1);
        float dv   = ldinv[node];
        float v    = lagg[i] + lh[i] * dv * dv + bias[ff];
        out[(size_t)(base + node) * FDIM + ff] = fmaxf(v, 0.0f);
    }
}

// ---------------- tiny final linear: [256,512] @ [512,10] + b ----------------
__global__ void k_lin2(const float* __restrict__ h, const float* __restrict__ w,
                       const float* __restrict__ b, float* __restrict__ out) {
    int i = blockIdx.x * blockDim.x + threadIdx.x;   // 2560 threads exactly
    if (i >= B_GRAPHS * OUT_DIM) return;
    int g = i / OUT_DIM;
    int o = i - g * OUT_DIM;
    float s = b[o];
    const float* hr = h + (size_t)g * MLP_H;
    for (int k = 0; k < MLP_H; ++k)
        s += hr[k] * w[k * OUT_DIM + o];
    out[i] = s;
}

// ---------------- launcher ----------------
extern "C" void kernel_launch(void* const* d_in, const int* in_sizes, int n_in,
                              void* d_out, int out_size, void* d_ws, size_t ws_size,
                              hipStream_t stream) {
    const float* x      = (const float*)d_in[0];
    const int*   eidx   = (const int*)  d_in[1];
    const int    E      = in_sizes[1] / 2;
    const int*   src    = eidx;
    const int*   dst    = eidx + E;
    const float* W1     = (const float*)d_in[3];
    const float* b1     = (const float*)d_in[4];
    const float* W2     = (const float*)d_in[5];
    const float* b2     = (const float*)d_in[6];
    const float* lin1_w = (const float*)d_in[7];
    const float* lin1_b = (const float*)d_in[8];
    const float* lin2_w = (const float*)d_in[9];
    const float* lin2_b = (const float*)d_in[10];
    float*       outp   = (float*)d_out;

    // workspace carve-up (floats): dinv | hbuf | act | mlp1  (~34.2 MB)
    float* dinv = (float*)d_ws;
    float* hbuf = dinv + N_TOT;
    float* act  = hbuf + (size_t)N_TOT * FDIM;
    float* mlp1 = act  + (size_t)N_TOT * FDIM;

    // 1) symmetric-normalization coefficients
    k_deg_init <<<N_TOT / 256, 256, 0, stream>>>(dinv);
    k_deg_accum<<<(E + 255) / 256, 256, 0, stream>>>(dst, dinv, E);
    k_rsqrt    <<<N_TOT / 256, 256, 0, stream>>>(dinv);

    // 2) layer 1: GEMM (32768x128 @ 128x128) -> LDS aggregation
    {
        int blocks = (N_TOT / (16 * 8)) * (FDIM / NBLK);   // 256*2 = 512
        k_wmma_gemm<8, false, false><<<blocks, 8 * 32, 0, stream>>>(x, W1, nullptr, hbuf,
                                                                    N_TOT, FDIM, FDIM);
        k_gcn_agg<<<B_GRAPHS, 256, 0, stream>>>(src, dst, dinv, hbuf, b1, act);
    }

    // 3) layer 2
    {
        int blocks = (N_TOT / (16 * 8)) * (FDIM / NBLK);
        k_wmma_gemm<8, false, false><<<blocks, 8 * 32, 0, stream>>>(act, W2, nullptr, hbuf,
                                                                    N_TOT, FDIM, FDIM);
        k_gcn_agg<<<B_GRAPHS, 256, 0, stream>>>(src, dst, dinv, hbuf, b2, act);
    }

    // 4) MLP lin1: [256,16384] @ [16384,512] + bias, relu (act is the flat view)
    {
        int blocks = (B_GRAPHS / (16 * 2)) * (MLP_H / NBLK);   // 8*8 = 64
        k_wmma_gemm<2, true, true><<<blocks, 2 * 32, 0, stream>>>(act, lin1_w, lin1_b, mlp1,
                                                                  B_GRAPHS, MLP_H, KDIM_MLP);
    }

    // 5) lin2 -> output [256,10]
    k_lin2<<<(B_GRAPHS * OUT_DIM) / 256, 256, 0, stream>>>(mlp1, lin2_w, lin2_b, outp);
}
